// LMLIIF_78134045049275
// MI455X (gfx1250) — matrix operand order repaired and checked by reading, not compiled, not run on hardware
//
#include <hip/hip_runtime.h>
#include <hip/hip_bf16.h>
#include <stdint.h>

typedef __attribute__((ext_vector_type(16))) _Float16 v16h;
typedef __attribute__((ext_vector_type(8)))  float    v8f;
typedef __attribute__((ext_vector_type(4)))  unsigned u32x4;
typedef __attribute__((ext_vector_type(8)))  unsigned u32x8;

#define H_FEAT 96
#define W_FEAT 96
#define C_FEAT 64
#define NPIX   (H_FEAT * W_FEAT)         // 9216
#define NQ     (H_FEAT * 4 * W_FEAT * 4) // 147456
#define HID    64
#define MODD   256
#define KHYP   578
#define KHYP_P 608                       // 19 * 32
#define K0     68
#define K0_P   96                        // 3 * 32

// workspace layout (bytes, 256-aligned)
#define WS_MOD    0                          // NPIX*MODD f16   = 4,718,592
#define WS_WH1T   4718592                    // 256*608  f16   =   311,296
#define WS_WH2T   (WS_WH1T + 311296)         // 256*256  f16   =   131,072
#define WS_W0T    (WS_WH2T + 131072)         // 64*96    f16   =    12,288
#define WS_W1T    (WS_W0T + 12288)           // 64*64    f16   =     8,192
#define WS_W2T    (WS_W1T + 8192)            // 16*64    f16   =     2,048

#define WMMA_F16(A, B, C) \
  __builtin_amdgcn_wmma_f32_16x16x32_f16(false, (A), false, (B), (short)0, (C), false, false)

// A fragment (16x32 f16) from an LDS row-major tile.
// A k order used here (contiguous): lane L holds row M=L&15; K = (L>>4)*16 + i, contiguous 16
// (same k order as the B fragments).  -> two contiguous 16B runs => 2x ds_load_b128.
__device__ __forceinline__ v16h frag_a_lds(const _Float16* row0, int ldk, int kc) {
  const int lane = threadIdx.x & 31;
  const _Float16* p = row0 + (lane & 15) * ldk + kc + ((lane >> 4) << 4);
  v16h a;
#pragma unroll
  for (int i = 0; i < 8; ++i) { a[i] = p[i]; a[i + 8] = p[8 + i]; }
  return a;
}

// B fragment (32x16 f16) from n-major (transposed) f16 weights in global memory.
// B k order used here (contiguous): lane L holds col N=L&15; K = (L>>4)*16 + i, contiguous 16.
__device__ __forceinline__ v16h frag_b_glb(const _Float16* w, int ldk, int n0, int kc) {
  const int lane = threadIdx.x & 31;
  const _Float16* p = w + (n0 + (lane & 15)) * ldk + kc + ((lane >> 4) << 4);
  v16h b;
#pragma unroll
  for (int i = 0; i < 16; ++i) b[i] = p[i];
  return b;
}

// B fragment from an LDS-resident 256x32 f16 chunk (row n, 32 k contiguous).
__device__ __forceinline__ v16h frag_b_lds(const _Float16* buf, int n0) {
  const int lane = threadIdx.x & 31;
  const _Float16* p = buf + (n0 + (lane & 15)) * 32 + ((lane >> 4) << 4);
  v16h b;
#pragma unroll
  for (int i = 0; i < 16; ++i) b[i] = p[i];
  return b;
}

// Tensor Data Mover: 2D tile (tile_d0 x tile_d1) of f16, global -> LDS.
// Builds D# groups 0/1 per CDNA5 ISA and issues the
// 2-operand (groups 0+1, tensors <= 2D) form. Tracked by TENSORcnt.
__device__ __forceinline__ void tdm_load_2d_f16(unsigned lds_off, const void* gaddr,
                                                unsigned tile_d0, unsigned tile_d1,
                                                unsigned tensor_d0, unsigned tensor_d1,
                                                unsigned stride_d0) {
  unsigned long long ga = (unsigned long long)(uintptr_t)gaddr;
  u32x4 g0;
  g0[0] = 1u;                                            // count=1 (valid), user mode
  g0[1] = lds_off;                                       // lds_addr (bytes)
  g0[2] = (unsigned)ga;                                  // global_addr[31:0]
  g0[3] = (unsigned)((ga >> 32) & 0x01FFFFFFu) | 0x80000000u; // addr[56:32] | type=2
  u32x8 g1;
  g1[0] = 0x00010000u;                                   // wg_mask=0, data_size=1 (2B)
  g1[1] = (tensor_d0 & 0xFFFFu) << 16;                   // tensor_dim0[15:0]
  g1[2] = (tensor_d0 >> 16) | ((tensor_d1 & 0xFFFFu) << 16); // dim0 hi | dim1 lo
  g1[3] = (tensor_d1 >> 16) | (tile_d0 << 16);           // dim1 hi | tile_dim0
  g1[4] = tile_d1 & 0xFFFFu;                             // tile_dim1 (tile_dim2=0)
  g1[5] = stride_d0;                                     // tensor_dim0_stride[31:0]
  g1[6] = 0u;                                            // stride0 hi | stride1 lo
  g1[7] = 0u;                                            // stride1 hi
  asm volatile("tensor_load_to_lds %0, %1" :: "s"(g0), "s"(g1) : "memory");
}

// ---------------------------------------------------------------------------
// Kernel 0: transpose + f16-convert + zero-pad all weight matrices into ws.
// ---------------------------------------------------------------------------
__global__ void prep_weights(const float* __restrict__ Wh1, const float* __restrict__ Wh2,
                             const float* __restrict__ W0, const float* __restrict__ W1,
                             const float* __restrict__ W2,
                             _Float16* __restrict__ wh1t, _Float16* __restrict__ wh2t,
                             _Float16* __restrict__ w0t, _Float16* __restrict__ w1t,
                             _Float16* __restrict__ w2t) {
  const int stride = gridDim.x * blockDim.x;
  const int tid = blockIdx.x * blockDim.x + threadIdx.x;
  for (int i = tid; i < 256 * KHYP_P; i += stride) {
    int n = i / KHYP_P, k = i - n * KHYP_P;
    wh1t[i] = (k < KHYP) ? (_Float16)Wh1[k * 256 + n] : (_Float16)0.f;
  }
  for (int i = tid; i < 256 * 256; i += stride) {
    int n = i >> 8, k = i & 255;
    wh2t[i] = (_Float16)Wh2[k * 256 + n];
  }
  for (int i = tid; i < HID * K0_P; i += stride) {
    int n = i / K0_P, k = i - n * K0_P;
    w0t[i] = (k < K0) ? (_Float16)W0[k * HID + n] : (_Float16)0.f;
  }
  for (int i = tid; i < HID * HID; i += stride) {
    int n = i >> 6, k = i & 63;
    w1t[i] = (_Float16)W1[k * HID + n];
  }
  for (int i = tid; i < 16 * HID; i += stride) {
    int n = i >> 6, k = i & 63;
    w2t[i] = (n < 3) ? (_Float16)W2[k * 3 + n] : (_Float16)0.f;
  }
}

// ---------------------------------------------------------------------------
// Kernel 1: hypernet. Block = 16 latent pixels x 256 outputs, 4 waves,
// each wave owns 16x64 via 4 WMMA fragments. Layer-1 weights are streamed
// chunk-by-chunk into LDS by the Tensor Data Mover, double-buffered.
// ---------------------------------------------------------------------------
__global__ __launch_bounds__(128)
void hyper_mod(const float* __restrict__ feat, const float* __restrict__ cell,
               const _Float16* __restrict__ wh1t, const float* __restrict__ bh1,
               const _Float16* __restrict__ wh2t, const float* __restrict__ bh2,
               _Float16* __restrict__ mod) {
  __shared__ __align__(16) _Float16 Atile[16 * KHYP_P];   // 19,456 B
  __shared__ __align__(16) _Float16 Htile[16 * 256];      //  8,192 B
  __shared__ __align__(16) _Float16 Bbuf[2][256 * 32];    // 32,768 B (TDM dbl-buf)
  const int tid = threadIdx.x;
  const int pbase = blockIdx.x * 16;
  const int wave = tid >> 5, lane = tid & 31;

  const unsigned ldsB[2] = {(unsigned)(uintptr_t)&Bbuf[0][0],
                            (unsigned)(uintptr_t)&Bbuf[1][0]};
  // Prologue: kick off chunk 0 of Wh1^T (256 n-rows x 32 k) while building A.
  if (wave == 0)
    tdm_load_2d_f16(ldsB[0], wh1t, 32, 256, KHYP_P, 256, KHYP_P);

  // Build 3x3-unfold input tile in LDS (channel-major patch order: k = c*9 + dy*3+dx)
  for (int idx = tid; idx < 16 * KHYP_P; idx += 128) {
    int r = idx / KHYP_P, k = idx - r * KHYP_P;
    int p = pbase + r, y = p / W_FEAT, x = p - y * W_FEAT;
    float v = 0.f;
    if (k < 576) {
      int c = k / 9, t = k - 9 * c;
      int yy = y + t / 3 - 1, xx = x + t % 3 - 1;
      if (yy >= 0 && yy < H_FEAT && xx >= 0 && xx < W_FEAT)
        v = feat[c * NPIX + yy * W_FEAT + xx];
    } else if (k == 576) v = cell[2 * p]     * (float)H_FEAT;
    else if   (k == 577) v = cell[2 * p + 1] * (float)W_FEAT;
    Atile[idx] = (_Float16)v;
  }
  __syncthreads();

  const int n0 = wave * 64;
  const int nlo = lane & 15, mb = (lane >> 4) << 3;

  // Layer 1: (16x608) @ (608x256); B chunks arrive via TDM, one DMA in flight.
  v8f acc[4] = {};
#pragma unroll 1
  for (int c = 0; c < KHYP_P / 32; ++c) {
    if (wave == 0)
      __builtin_amdgcn_s_wait_tensorcnt(0);   // chunk c landed in Bbuf[c&1]
    __syncthreads();                           // publish buffer to all waves
    if (wave == 0 && c + 1 < KHYP_P / 32)      // overlap DMA of c+1 with compute of c
      tdm_load_2d_f16(ldsB[(c + 1) & 1], wh1t + (c + 1) * 32,
                      32, 256, KHYP_P, 256, KHYP_P);
    const _Float16* bb = Bbuf[c & 1];
    v16h a = frag_a_lds(Atile, KHYP_P, 32 * c);
#pragma unroll
    for (int f = 0; f < 4; ++f)
      acc[f] = WMMA_F16(a, frag_b_lds(bb, n0 + 16 * f), acc[f]);
  }
#pragma unroll
  for (int f = 0; f < 4; ++f) {
    int n = n0 + 16 * f + nlo;
    float bias = bh1[n];
#pragma unroll
    for (int r = 0; r < 8; ++r) {
      float v = acc[f][r] + bias;
      Htile[(mb + r) * 256 + n] = (_Float16)(v > 0.f ? v : 0.f);
    }
  }
  __syncthreads();  // layer-2 A reads columns written by other waves

  // Layer 2: (16x256) @ (256x256), B streamed from L2 via global_load_b128.
  v8f acc2[4] = {};
#pragma unroll 1
  for (int kc = 0; kc < 256; kc += 32) {
    v16h a = frag_a_lds(Htile, 256, kc);
#pragma unroll
    for (int f = 0; f < 4; ++f) {
      v16h b = frag_b_glb(wh2t, 256, n0 + 16 * f, kc);
      acc2[f] = WMMA_F16(a, b, acc2[f]);
    }
  }
#pragma unroll
  for (int f = 0; f < 4; ++f) {
    int n = n0 + 16 * f + nlo;
    float bias = bh2[n];
#pragma unroll
    for (int r = 0; r < 8; ++r)
      mod[(pbase + mb + r) * MODD + n] = (_Float16)(acc2[f][r] + bias);
  }
}

// ---------------------------------------------------------------------------
// Kernel 2: query_rgb. Block = 16 queries x 4 ensemble branches = 64 rows.
// 4 waves, each owning a private 16-row strip x 64 cols. Three WMMA layers
// with per-row modulation gathered from the hypernet output, then area blend.
// ---------------------------------------------------------------------------
__global__ __launch_bounds__(128)
void query_rgb(const float* __restrict__ feat, const float* __restrict__ coord,
               const float* __restrict__ cell, const _Float16* __restrict__ mod,
               const _Float16* __restrict__ w0t, const float* __restrict__ b0,
               const _Float16* __restrict__ w1t, const float* __restrict__ b1,
               const _Float16* __restrict__ w2t, const float* __restrict__ b2,
               float* __restrict__ out) {
  __shared__ __align__(16) _Float16 Atile[64 * K0_P];  // 12,288 B
  __shared__ __align__(16) _Float16 Htile[64 * HID];   //  8,192 B
  __shared__ float Pred[64 * 4];
  __shared__ float Area[64], RelY[64], RelX[64];
  __shared__ int   Pidx[64];
  const int tid = threadIdx.x;
  const int qbase = blockIdx.x * 16;

  __builtin_prefetch(w0t, 0, 3);  // global_prefetch_b8 on the hot weight stream
  __builtin_prefetch(w1t, 0, 3);

  if (tid < 64) {
    int r = tid, q = qbase + (r >> 2), b = r & 3;
    float vx = (b < 2) ? -1.f : 1.f;           // applies to y
    float vy = (b & 1) ? 1.f : -1.f;           // applies to x
    float cy = coord[2 * q], cx = coord[2 * q + 1];
    float gy = cy + vx * (1.f / H_FEAT) + 1e-6f;
    float gx = cx + vy * (1.f / W_FEAT) + 1e-6f;
    gy = fminf(fmaxf(gy, -1.f + 1e-6f), 1.f - 1e-6f);
    gx = fminf(fmaxf(gx, -1.f + 1e-6f), 1.f - 1e-6f);
    int iy = (int)fminf(fmaxf(roundf((gy + 1.f) * (H_FEAT * 0.5f) - 0.5f), 0.f), (float)(H_FEAT - 1));
    int ix = (int)fminf(fmaxf(roundf((gx + 1.f) * (W_FEAT * 0.5f) - 0.5f), 0.f), (float)(W_FEAT - 1));
    Pidx[r] = iy * W_FEAT + ix;
    float qy = ((iy + 0.5f) / H_FEAT) * 2.f - 1.f;
    float qx = ((ix + 0.5f) / W_FEAT) * 2.f - 1.f;
    float ry = (cy - qy) * (float)H_FEAT;
    float rx = (cx - qx) * (float)W_FEAT;
    RelY[r] = ry; RelX[r] = rx;
    Area[r] = fabsf(ry * rx) + 1e-9f;
  }
  __syncthreads();

  // Build A tile: [feat gather (64) | rel_y rel_x | rel_cell_y rel_cell_x | zero pad]
  for (int idx = tid; idx < 64 * K0_P; idx += 128) {
    int r = idx / K0_P, k = idx - r * K0_P;
    float v = 0.f;
    if (k < 64)        v = feat[k * NPIX + Pidx[r]];
    else if (k == 64)  v = RelY[r];
    else if (k == 65)  v = RelX[r];
    else if (k == 66)  v = cell[2 * (qbase + (r >> 2))]     * (float)H_FEAT;
    else if (k == 67)  v = cell[2 * (qbase + (r >> 2)) + 1] * (float)W_FEAT;
    Atile[idx] = (_Float16)v;
  }
  __syncthreads();

  const int wave = tid >> 5, lane = tid & 31;
  const int rbase = wave * 16;
  const int nlo = lane & 15, mb = (lane >> 4) << 3;

  // Layer 0: (16x96) @ (96x64)  (K zero-padded 68->96)
  v8f acc[4] = {};
#pragma unroll
  for (int kc = 0; kc < K0_P; kc += 32) {
    v16h a = frag_a_lds(Atile + rbase * K0_P, K0_P, kc);
#pragma unroll
    for (int f = 0; f < 4; ++f) {
      v16h b = frag_b_glb(w0t, K0_P, 16 * f, kc);
      acc[f] = WMMA_F16(a, b, acc[f]);
    }
  }
#pragma unroll
  for (int f = 0; f < 4; ++f) {
    int n = 16 * f + nlo;
    float bias = b0[n];
#pragma unroll
    for (int r = 0; r < 8; ++r) {
      int m = rbase + mb + r;
      const _Float16* mp = mod + Pidx[m] * MODD;
      float v = (acc[f][r] + bias) * (float)mp[n] + (float)mp[128 + n];
      Htile[m * HID + n] = (_Float16)(v > 0.f ? v : 0.f);
    }
  }
  // (no barrier: each wave's 16-row Htile strip is private; LDS is in-order per wave)

  // Layer 1: (16x64) @ (64x64)
  v8f acc2[4] = {};
#pragma unroll
  for (int kc = 0; kc < HID; kc += 32) {
    v16h a = frag_a_lds(Htile + rbase * HID, HID, kc);
#pragma unroll
    for (int f = 0; f < 4; ++f) {
      v16h b = frag_b_glb(w1t, HID, 16 * f, kc);
      acc2[f] = WMMA_F16(a, b, acc2[f]);
    }
  }
#pragma unroll
  for (int f = 0; f < 4; ++f) {
    int n = 16 * f + nlo;
    float bias = b1[n];
#pragma unroll
    for (int r = 0; r < 8; ++r) {
      int m = rbase + mb + r;
      const _Float16* mp = mod + Pidx[m] * MODD;
      float v = (acc2[f][r] + bias) * (float)mp[64 + n] + (float)mp[192 + n];
      Htile[m * HID + n] = (_Float16)(v > 0.f ? v : 0.f);
    }
  }

  // Layer 2: (16x64) @ (64x16), only cols 0..2 valid (W2 zero-padded)
  v8f acc3 = {};
#pragma unroll
  for (int kc = 0; kc < HID; kc += 32) {
    v16h a = frag_a_lds(Htile + rbase * HID, HID, kc);
    v16h b = frag_b_glb(w2t, HID, 0, kc);
    acc3 = WMMA_F16(a, b, acc3);
  }
  if (nlo < 3) {
#pragma unroll
    for (int r = 0; r < 8; ++r)
      Pred[(rbase + mb + r) * 4 + nlo] = acc3[r] + b2[nlo];
  }
  __syncthreads();

  // LIIF local-ensemble blend with diagonal area swap (0<->3, 1<->2)
  if (tid < 48) {
    int ql = tid / 3, c = tid - 3 * ql;
    int rb = 4 * ql;
    float a0 = Area[rb + 0], a1 = Area[rb + 1], a2 = Area[rb + 2], a3 = Area[rb + 3];
    float tot = a0 + a1 + a2 + a3;
    float v = Pred[(rb + 0) * 4 + c] * a3 + Pred[(rb + 1) * 4 + c] * a2 +
              Pred[(rb + 2) * 4 + c] * a1 + Pred[(rb + 3) * 4 + c] * a0;
    out[(qbase + ql) * 3 + c] = v / tot;
  }
}

extern "C" void kernel_launch(void* const* d_in, const int* in_sizes, int n_in,
                              void* d_out, int out_size, void* d_ws, size_t ws_size,
                              hipStream_t stream) {
  const float* feat  = (const float*)d_in[0];
  const float* coord = (const float*)d_in[1];
  const float* cell  = (const float*)d_in[2];
  const float* Wh1   = (const float*)d_in[3];
  const float* bh1   = (const float*)d_in[4];
  const float* Wh2   = (const float*)d_in[5];
  const float* bh2   = (const float*)d_in[6];
  const float* W0    = (const float*)d_in[7];
  const float* b0    = (const float*)d_in[8];
  const float* W1    = (const float*)d_in[9];
  const float* b1    = (const float*)d_in[10];
  const float* W2    = (const float*)d_in[11];
  const float* b2    = (const float*)d_in[12];

  char* ws = (char*)d_ws;
  _Float16* modb = (_Float16*)(ws + WS_MOD);
  _Float16* wh1t = (_Float16*)(ws + WS_WH1T);
  _Float16* wh2t = (_Float16*)(ws + WS_WH2T);
  _Float16* w0t  = (_Float16*)(ws + WS_W0T);
  _Float16* w1t  = (_Float16*)(ws + WS_W1T);
  _Float16* w2t  = (_Float16*)(ws + WS_W2T);

  prep_weights<<<256, 256, 0, stream>>>(Wh1, Wh2, W0, W1, W2,
                                        wh1t, wh2t, w0t, w1t, w2t);
  hyper_mod<<<NPIX / 16, 128, 0, stream>>>(feat, cell, wh1t, bh1, wh2t, bh2, modb);
  query_rgb<<<NQ / 16, 128, 0, stream>>>(feat, coord, cell, modb,
                                         w0t, b0, w1t, b1, w2t, b2,
                                         (float*)d_out);
}
